// HouseholderFlow_80315888436082
// MI455X (gfx1250) — matrix operand order repaired and checked by reading, not compiled, not run on hardware
//
#include <hip/hip_runtime.h>

// ---------------- problem constants ----------------
#define NUM_FLOW 8
#define HIDDEN   1024
#define CODE     512
#define BATCH    16384
#define TM       64              // batch rows per block

// ---------------- LDS layout (bytes) ----------------
#define ASTRIDE      520         // ushorts per abuf row (512 + 8 pad -> 1040B, 16B bank shift)
#define ZSTRIDE      512         // floats per zbuf row (row-wise access: conflict-free)
#define BSTRIDE      40          // ushorts per bbuf row (64B data + 16B pad, keeps 16B alignment)
#define ABUF_BYTES   (TM * ASTRIDE * 2)          // 66560
#define ZBUF_OFF     ABUF_BYTES
#define ZBUF_BYTES   (TM * ZSTRIDE * 4)          // 131072
#define BBUF_OFF     (ZBUF_OFF + ZBUF_BYTES)     // 197632
#define BHALF_BYTES  (CODE * BSTRIDE * 2)        // 40960 (one 512 x 32 bf16 K-chunk, padded)
#define SMEM_BYTES   (BBUF_OFF + 2 * BHALF_BYTES) // 279552 < 320KB/WGP

typedef __bf16 v16bf __attribute__((ext_vector_type(16)));
typedef float  v8f   __attribute__((ext_vector_type(8)));
typedef unsigned short ushort_t;
typedef ushort_t us8 __attribute__((ext_vector_type(8)));
typedef ushort_t us4 __attribute__((ext_vector_type(4)));
typedef unsigned int u32x4 __attribute__((ext_vector_type(4)));
typedef int i32x8 __attribute__((ext_vector_type(8)));
typedef int i32x4 __attribute__((ext_vector_type(4)));

static __device__ __forceinline__ ushort_t f2bf(float f) {
  unsigned u = __builtin_bit_cast(unsigned, f);
  u += 0x7FFFu + ((u >> 16) & 1u);       // round-to-nearest-even
  return (ushort_t)(u >> 16);
}
static __device__ __forceinline__ float bf2f(ushort_t h) {
  return __builtin_bit_cast(float, ((unsigned)h) << 16);
}

union FragAB { us8 u[2]; v16bf v; };

// ---------------- pre-pass: f32 weights -> bf16 in d_ws ----------------
// ws layout: W0 [512][1024] then Ws[6][512][512], row-major [N][K]
__global__ void cvt_weights(const float* __restrict__ W0,
                            const float* __restrict__ Ws,
                            ushort_t* __restrict__ out) {
  const int NW0 = CODE * HIDDEN;
  int i4   = blockIdx.x * blockDim.x + threadIdx.x;
  int base = i4 * 4;
  float4 w;
  if (base < NW0) w = ((const float4*)W0)[i4];
  else            w = ((const float4*)Ws)[i4 - NW0 / 4];
  us4 o;
  o[0] = f2bf(w.x); o[1] = f2bf(w.y); o[2] = f2bf(w.z); o[3] = f2bf(w.w);
  *(us4*)(out + base) = o;
}

// ---------------- TDM: DMA one 512-row x 64B K-chunk of W into a bbuf half ----
// 2D descriptor: tile_dim0 = 32 elems (64B), tile_dim1 = 512 rows,
// row stride in memory = Kl elems; LDS padding 4 DWORDs every 16 DWORDs -> 80B rows.
static __device__ __forceinline__ void tdm_issue(const ushort_t* W, int Kl, int k0,
                                                 int buf) {
  unsigned long long ga = (unsigned long long)(const void*)(W + k0);
  u32x4 g0;
  g0[0] = 1u;                                          // count=1, user descriptor
  g0[1] = (unsigned)(BBUF_OFF + buf * BHALF_BYTES);    // lds_addr (bytes)
  g0[2] = (unsigned)(ga & 0xFFFFFFFFu);                // global_addr[31:0]
  g0[3] = (unsigned)((ga >> 32) & 0x01FFFFFFu) | (2u << 30);   // addr[56:32] | type=2
  i32x8 g1;
  g1[0] = (int)((1u << 16)        // data_size = 2B
              | (1u << 20)        // pad_enable
              | (3u << 22)        // pad_interval = 16 DWORDs (64B)
              | (3u << 25));      // pad_amount   = 4 DWORDs (16B)
  g1[1] = (int)(((unsigned)Kl & 0xFFFFu) << 16);       // tensor_dim0[15:0] @ bits 63:48
  g1[2] = (int)((((unsigned)Kl >> 16) & 0xFFFFu)       // tensor_dim0[31:16]
              | (512u << 16));                          // tensor_dim1[15:0]
  g1[3] = (int)(32u << 16);                             // tile_dim0 = 32 elems
  g1[4] = 512;                                          // tile_dim1 = 512 rows
  g1[5] = Kl;                                           // tensor_dim0_stride[31:0]
  g1[6] = 0;
  g1[7] = 0;
  i32x4 z4 = {0, 0, 0, 0};                              // groups 2/3 unused (2D)
  i32x8 z8 = {0, 0, 0, 0, 0, 0, 0, 0};
  __builtin_amdgcn_tensor_load_to_lds(g0, g1, z4, z4, z8, 0);  // 6-arg toolchain form
}

// ---------------- fused 7-layer Householder flow ----------------
__global__ __launch_bounds__(256, 1)
void hhflow_kernel(const float* __restrict__ hidden,
                   const float* __restrict__ zs,
                   const float* __restrict__ b0,
                   const float* __restrict__ bs,
                   const ushort_t* __restrict__ wbf,
                   float* __restrict__ out) {
  extern __shared__ char smem_raw[];
  ushort_t* abuf = (ushort_t*)smem_raw;                    // [64][520] bf16 activations
  float*    zbuf = (float*)(smem_raw + ZBUF_OFF);          // [64][512] f32 z
  ushort_t* bbuf = (ushort_t*)(smem_raw + BBUF_OFF);       // [2][512][40] bf16 W chunk

  const int tid = threadIdx.x;
  const int m0  = blockIdx.x * TM;

  // ---- stage zs into LDS ----
  {
    const float4* zp = (const float4*)(zs + (size_t)m0 * CODE);
    for (int i = 0; i < 32; ++i) {
      int e = i * 256 + tid;
      int row = e >> 7, c4 = e & 127;
      *(float4*)(zbuf + row * ZSTRIDE + c4 * 4) = zp[row * 128 + c4];
    }
  }
  __syncthreads();

  const int lane = tid & 31;
  const int wid  = tid >> 5;
  const int l16  = lane & 15;
  const int hi   = lane >> 4;       // lane half selects K sub-chunk (ISA A/B layouts)
  const int m_w  = wid & 1;         // 2 waves tile M (32 rows each)
  const int n_w  = wid >> 1;        // 4 waves tile N (128 cols each)

  for (int layer = 0; layer < NUM_FLOW - 1; ++layer) {
    const int Kl = (layer == 0) ? HIDDEN : CODE;
    const ushort_t* W = (layer == 0)
        ? wbf
        : (wbf + (size_t)CODE * HIDDEN + (size_t)(layer - 1) * CODE * CODE);
    const float* bias = (layer == 0) ? b0 : (bs + (layer - 1) * CODE);

    v8f acc[2][8];
#pragma unroll
    for (int mt = 0; mt < 2; ++mt)
#pragma unroll
      for (int nt = 0; nt < 8; ++nt)
#pragma unroll
        for (int j = 0; j < 8; ++j) acc[mt][nt][j] = 0.0f;

    const int nphase = (layer == 0) ? 2 : 1;
    for (int kh = 0; kh < nphase; ++kh) {
      if (layer == 0) {
        // stage 512-wide K-half of hidden as bf16 into abuf
        __syncthreads();                 // prior abuf readers done
        const float4* hp = (const float4*)(hidden + (size_t)m0 * HIDDEN + kh * CODE);
        for (int i = 0; i < 32; ++i) {
          int e = i * 256 + tid;
          int row = e >> 7, c4 = e & 127;
          float4 h = hp[row * 256 + c4];       // row stride = 1024 f32 = 256 float4
          us4 o;
          o[0] = f2bf(h.x); o[1] = f2bf(h.y); o[2] = f2bf(h.z); o[3] = f2bf(h.w);
          *(us4*)(abuf + row * ASTRIDE + c4 * 4) = o;
        }
      }

      // ---- TDM double-buffered B pipeline over 16 K-chunks of 32 ----
      if (wid == 0) tdm_issue(W, Kl, kh * CODE, 0);
      for (int c = 0; c < 16; ++c) {
        const int buf = c & 1;
        if (wid == 0) __builtin_amdgcn_s_wait_tensorcnt((short)0);
        __syncthreads();   // publish DMA'd tile; prior readers of other buf done
        if (wid == 0 && c + 1 < 16) tdm_issue(W, Kl, kh * CODE + (c + 1) * 32, buf ^ 1);

        const int klocal = c * 32;
        FragAB a[2];
#pragma unroll
        for (int mt = 0; mt < 2; ++mt) {
          const ushort_t* ar =
              abuf + (m_w * 32 + mt * 16 + l16) * ASTRIDE + klocal + hi * 8;
          a[mt].u[0] = *(const us8*)(ar);
          a[mt].u[1] = *(const us8*)(ar + 16);
        }
        const ushort_t* bb =
            bbuf + buf * (BHALF_BYTES / 2) + (n_w * 128 + l16) * BSTRIDE + hi * 8;
#pragma unroll
        for (int nt = 0; nt < 8; ++nt) {
          FragAB b;
          const ushort_t* bp = bb + nt * 16 * BSTRIDE;
          b.u[0] = *(const us8*)(bp);
          b.u[1] = *(const us8*)(bp + 16);
          acc[0][nt] = __builtin_amdgcn_wmma_f32_16x16x32_bf16(
              false, a[0].v, false, b.v, (short)0, acc[0][nt], false, false);
          acc[1][nt] = __builtin_amdgcn_wmma_f32_16x16x32_bf16(
              false, a[1].v, false, b.v, (short)0, acc[1][nt], false, false);
        }
      }
    }

    // ---- bias ----
#pragma unroll
    for (int nt = 0; nt < 8; ++nt) {
      float bv = bias[n_w * 128 + nt * 16 + l16];
      acc[0][nt] += bv;
      acc[1][nt] += bv;
    }

    __syncthreads();   // all waves done reading abuf

    // ---- store v as bf16 into abuf (next layer's A / HH input) ----
#pragma unroll
    for (int mt = 0; mt < 2; ++mt)
#pragma unroll
      for (int nt = 0; nt < 8; ++nt) {
        const int n = n_w * 128 + nt * 16 + l16;
#pragma unroll
        for (int r = 0; r < 8; ++r) {
          abuf[(m_w * 32 + mt * 16 + hi * 8 + r) * ASTRIDE + n] = f2bf(acc[mt][nt][r]);
        }
      }
    __syncthreads();

    // ---- Householder: each wave owns 8 rows exclusively ----
    for (int rr = 0; rr < 8; ++rr) {
      const int m = wid * 8 + rr;
      const ushort_t* vr = abuf + m * ASTRIDE + lane * 16;
      float*          zr = zbuf + m * ZSTRIDE + lane * 16;
      us8 a0 = *(const us8*)vr;
      us8 a1 = *(const us8*)(vr + 8);
      float v[16];
#pragma unroll
      for (int j = 0; j < 8; ++j) { v[j] = bf2f(a0[j]); v[8 + j] = bf2f(a1[j]); }
      float4 zq[4];
#pragma unroll
      for (int q = 0; q < 4; ++q) zq[q] = *(const float4*)(zr + q * 4);
      const float* zf = (const float*)zq;
      float vz = 0.f, v2 = 0.f;
#pragma unroll
      for (int j = 0; j < 16; ++j) { vz += v[j] * zf[j]; v2 += v[j] * v[j]; }
#pragma unroll
      for (int off = 16; off >= 1; off >>= 1) {
        vz += __shfl_xor(vz, off, 32);
        v2 += __shfl_xor(v2, off, 32);
      }
      const float s = 2.0f * vz / v2;
#pragma unroll
      for (int q = 0; q < 4; ++q) {
        float4 o;
        o.x = zf[q * 4 + 0] - s * v[q * 4 + 0];
        o.y = zf[q * 4 + 1] - s * v[q * 4 + 1];
        o.z = zf[q * 4 + 2] - s * v[q * 4 + 2];
        o.w = zf[q * 4 + 3] - s * v[q * 4 + 3];
        *(float4*)(zr + q * 4) = o;
      }
    }
    // next iteration's barriers protect abuf/bbuf reuse
  }

  __syncthreads();
  // ---- write final z tile to HBM ----
  float4* op = (float4*)(out + (size_t)m0 * CODE);
  for (int i = 0; i < 32; ++i) {
    int e = i * 256 + tid;
    int row = e >> 7, c4 = e & 127;
    op[row * 128 + c4] = *(const float4*)(zbuf + row * ZSTRIDE + c4 * 4);
  }
}

// ---------------- launch ----------------
extern "C" void kernel_launch(void* const* d_in, const int* in_sizes, int n_in,
                              void* d_out, int out_size, void* d_ws, size_t ws_size,
                              hipStream_t stream) {
  const float* hidden = (const float*)d_in[0];
  const float* zs     = (const float*)d_in[1];
  const float* W0     = (const float*)d_in[2];
  const float* b0     = (const float*)d_in[3];
  const float* Ws     = (const float*)d_in[4];
  const float* bs     = (const float*)d_in[5];
  ushort_t*    wbf    = (ushort_t*)d_ws;   // 4 MB bf16 weights

  cvt_weights<<<2048, 256, 0, stream>>>(W0, Ws, wbf);
  hhflow_kernel<<<BATCH / TM, 256, SMEM_BYTES, stream>>>(hidden, zs, b0, bs, wbf,
                                                         (float*)d_out);
}
